// inputToLeafSinkhorn_28063316312541
// MI455X (gfx1250) — matrix operand order repaired and checked by reading, not compiled, not run on hardware
//
#include <hip/hip_runtime.h>
#include <hip/hip_bf16.h>
#include <math.h>

// ---------------------------------------------------------------------------
// inputToLeafSinkhorn for MI455X (gfx1250, wave32)
//
// Roofline: GEMM is 17.2 GFLOP vs 537 MB mandatory HBM traffic (AI ~32 F/B).
// Bandwidth-bound only if matrix throughput > ~745 TFLOP/s, which rules out
// the f32 WMMA path. We use split-precision f16x2 (hi/lo) WMMA with f32
// accumulation: 3 v_wmma_f32_16x16x32_f16 per tile -> near-f32 accuracy at
// f16 matrix speed, keeping the kernel HBM-bound (~23 us at 23.3 TB/s).
// x / out are streamed exactly once -> non-temporal loads/stores so the
// 537 MB stream does not evict the L2-resident P matrix (64 KB, read by all
// 4096 blocks).
// ---------------------------------------------------------------------------

typedef __attribute__((ext_vector_type(16))) _Float16 v16h;
typedef __attribute__((ext_vector_type(8)))  _Float16 v8h;
typedef __attribute__((ext_vector_type(8)))  float    v8f;
typedef __attribute__((ext_vector_type(4)))  float    v4f;

#define NN 128
#define A_PITCH 132   // floats: pad so row (float4) & column (scalar) LDS sweeps avoid bank conflicts
#define B_PITCH 136   // halfs:  pad so 16-lane B-tile ds_load_b128s hit distinct banks

// ---------------------------------------------------------------------------
// Kernel 1: Gumbel + Sinkhorn(20) -> P, split into f16 hi/lo, one workgroup.
// Scaling-vector form: r = 1/(A0 @ c); c = 1/(r @ A0); P = diag(r) A0 diag(c).
// ---------------------------------------------------------------------------
__global__ __launch_bounds__(128) void sinkhorn_split_kernel(
    const float* __restrict__ logits, const float* __restrict__ u,
    _Float16* __restrict__ Phi, _Float16* __restrict__ Plo)
{
  __shared__ float A0[NN * A_PITCH];
  __shared__ float rvec[NN];
  __shared__ float cvec[NN];
  const int t = threadIdx.x;   // thread t owns row t (row pass) and column t (col pass)

  // A0[t][k] = exp(clip((logits + gumbel)/3, -10, 10))
  for (int k = 0; k < NN; ++k) {
    float uu = u[t * NN + k];
    float g  = -logf(-logf(uu + 1e-20f) + 1e-20f);   // NOISE_SCALE = 1
    float la = (logits[t * NN + k] + g) / 3.0f;      // TEMPERATURE = 3
    la = fminf(10.0f, fmaxf(-10.0f, la));
    A0[t * A_PITCH + k] = expf(la);
  }
  cvec[t] = 1.0f;
  __syncthreads();

  for (int it = 0; it < 20; ++it) {
    // r = 1 / (A0 @ c)   (row sweep, vectorizable: pitch 132 keeps banks spread)
    float s = 0.0f;
    #pragma unroll 8
    for (int k = 0; k < NN; ++k) s += A0[t * A_PITCH + k] * cvec[k];
    rvec[t] = 1.0f / s;
    __syncthreads();
    // c = 1 / (r @ A0)   (column sweep: banks (4i + t) % 64 are conflict-free)
    float s2 = 0.0f;
    #pragma unroll 8
    for (int i = 0; i < NN; ++i) s2 += rvec[i] * A0[i * A_PITCH + t];
    cvec[t] = 1.0f / s2;
    __syncthreads();
  }

  // P = nan_to_num(diag(r) A0 diag(c)); split into f16 hi + f16 residual lo
  const float rt = rvec[t];
  for (int k = 0; k < NN; ++k) {
    float p = rt * A0[t * A_PITCH + k] * cvec[k];
    if (isnan(p))      p = 0.0f;
    else if (isinf(p)) p = (p > 0.0f) ? 1.0f : 0.0f;
    _Float16 hi = (_Float16)p;
    _Float16 lo = (_Float16)(p - (float)hi);
    Phi[t * NN + k] = hi;
    Plo[t * NN + k] = lo;
  }
}

// ---------------------------------------------------------------------------
// Kernel 2: out = x @ P.T via v_wmma_f32_16x16x32_f16, split-precision f16x2.
// Block = 256 threads (8 waves), 128 rows of x per block; wave w owns rows
// [blk*128 + 16w, +16) x all 128 columns = 8 accumulator tiles (64 VGPRs).
// P_hi/P_lo staged in LDS (padded pitch 136 halfs -> conflict-free b128 reads).
// ---------------------------------------------------------------------------
__global__ __launch_bounds__(256) void xPt_wmma_kernel(
    const float* __restrict__ x,
    const _Float16* __restrict__ Phi, const _Float16* __restrict__ Plo,
    float* __restrict__ out)
{
  __shared__ _Float16 Bh[NN * B_PITCH];
  __shared__ _Float16 Bl[NN * B_PITCH];
  const int tid = threadIdx.x;

  // Cooperative stage of P_hi/P_lo (32 KB each) into padded LDS, 16B chunks.
  // P is hot across all blocks -> default (RT) temporal hint keeps it in L2.
  for (int idx = tid; idx < NN * (NN / 8); idx += 256) {
    const int row = idx >> 4;
    const int col = (idx & 15) << 3;
    *(v8h*)&Bh[row * B_PITCH + col] = *(const v8h*)&Phi[row * NN + col];
    *(v8h*)&Bl[row * B_PITCH + col] = *(const v8h*)&Plo[row * NN + col];
  }
  __syncthreads();

  const int wave = tid >> 5;
  const int lane = tid & 31;          // wave32
  const int m    = lane & 15;         // A row within 16-tile / B column
  const int g    = lane >> 4;         // half-wave group (K-phase selector)
  const long rowBase = (long)blockIdx.x * 128 + (long)wave * 16;
  const float* __restrict__ xrow = x + (rowBase + m) * NN;

  v8f acc[8];
  #pragma unroll
  for (int nt = 0; nt < 8; ++nt) acc[nt] = (v8f){0,0,0,0,0,0,0,0};

  #pragma unroll
  for (int kc = 0; kc < 4; ++kc) {    // K = 128 in 4 chunks of 32
    // A fragment, 16-bit A layout (16x32): lane group g, halves 0..7 hold
    // K = 8g+32kc+0..7 ; halves 8..15 hold K = 16+8g+32kc+0..7.
    // x is streamed once -> non-temporal loads (TH=NT), keep L2 for P.
    const int k0 = kc * 32 + g * 8;
    v4f f0 = __builtin_nontemporal_load((const v4f*)(xrow + k0));
    v4f f1 = __builtin_nontemporal_load((const v4f*)(xrow + k0 + 4));
    v4f f2 = __builtin_nontemporal_load((const v4f*)(xrow + k0 + 16));
    v4f f3 = __builtin_nontemporal_load((const v4f*)(xrow + k0 + 20));
    float f[16];
    #pragma unroll
    for (int i = 0; i < 4; ++i) { f[i]=f0[i]; f[4+i]=f1[i]; f[8+i]=f2[i]; f[12+i]=f3[i]; }
    v16h ahi, alo;
    #pragma unroll
    for (int i = 0; i < 16; ++i) {
      _Float16 h = (_Float16)f[i];
      ahi[i] = h;
      alo[i] = (_Float16)(f[i] - (float)h);   // f16 residual: ~22-bit combined mantissa
    }

    #pragma unroll
    for (int nt = 0; nt < 8; ++nt) {
      // B layout (32x16): lane column = m, same K order as A: halves 0..7 hold
      // K = 32*kc + 8*g + 0..7, halves 8..15 hold K = 32*kc + 16 + 8*g + 0..7.
      const int boff = (nt * 16 + m) * B_PITCH + kc * 32 + g * 8;
      v8h bh0 = *(const v8h*)&Bh[boff];
      v8h bh1 = *(const v8h*)&Bh[boff + 16];
      v8h bl0 = *(const v8h*)&Bl[boff];
      v8h bl1 = *(const v8h*)&Bl[boff + 16];
      v16h bhi, blo;
      #pragma unroll
      for (int i = 0; i < 8; ++i) { bhi[i]=bh0[i]; bhi[8+i]=bh1[i]; blo[i]=bl0[i]; blo[8+i]=bl1[i]; }

      acc[nt] = __builtin_amdgcn_wmma_f32_16x16x32_f16(false, ahi, false, bhi,
                                                       (short)0, acc[nt], false, false);
      acc[nt] = __builtin_amdgcn_wmma_f32_16x16x32_f16(false, ahi, false, blo,
                                                       (short)0, acc[nt], false, false);
      acc[nt] = __builtin_amdgcn_wmma_f32_16x16x32_f16(false, alo, false, bhi,
                                                       (short)0, acc[nt], false, false);
    }
  }

  // D layout: lane L, VGPR v -> row = v + 8g, col = m + 16*nt.
  // out is written once -> non-temporal stores.
  #pragma unroll
  for (int nt = 0; nt < 8; ++nt) {
    #pragma unroll
    for (int v = 0; v < 8; ++v) {
      __builtin_nontemporal_store(acc[nt][v],
          &out[(rowBase + g * 8 + v) * NN + nt * 16 + m]);
    }
  }
}

// ---------------------------------------------------------------------------
extern "C" void kernel_launch(void* const* d_in, const int* in_sizes, int n_in,
                              void* d_out, int out_size, void* d_ws, size_t ws_size,
                              hipStream_t stream) {
  const float* x      = (const float*)d_in[0];   // [524288, 128] f32
  const float* logits = (const float*)d_in[1];   // [128, 128] f32
  const float* u      = (const float*)d_in[2];   // [128, 128] f32
  float* out = (float*)d_out;                    // [524288, 128] f32

  _Float16* Phi = (_Float16*)d_ws;               // 32 KB
  _Float16* Plo = Phi + NN * NN;                 // 32 KB

  sinkhorn_split_kernel<<<1, 128, 0, stream>>>(logits, u, Phi, Plo);

  const int M = in_sizes[0] / NN;                // 524288 rows
  xPt_wmma_kernel<<<M / 128, 256, 0, stream>>>(x, Phi, Plo, out);
}
